// NonLocalBlock_41764261987052
// MI455X (gfx1250) — matrix operand order repaired and checked by reading, not compiled, not run on hardware
//
#include <hip/hip_runtime.h>
#include <hip/hip_bf16.h>
#include <math.h>

#define Bn  4
#define Cn  128
#define HWn 4096

typedef __bf16 bf16_t;
typedef __attribute__((ext_vector_type(16))) __bf16 v16bf;
typedef __attribute__((ext_vector_type(8)))  __bf16 v8bf;
typedef __attribute__((ext_vector_type(8)))  float  v8f;

__device__ __forceinline__ bf16_t f2bf(float f) { return (bf16_t)f; }

__device__ __forceinline__ v16bf mk16(const bf16_t* plo, const bf16_t* phi) {
  v8bf lo = *(const v8bf*)plo;
  v8bf hi = *(const v8bf*)phi;
  v16bf r;
#pragma unroll
  for (int i = 0; i < 8; ++i) { r[i] = lo[i]; r[i + 8] = hi[i]; }
  return r;
}

// A fragment 16x32 bf16 from row-major [row][k] (k contiguous), pre-offset to (row0,k0).
__device__ __forceinline__ v16bf frag_a(const bf16_t* src, int ld, int half, int l16) {
  const bf16_t* p = src + (size_t)l16 * ld + half * 8;
  return mk16(p, p + 16);
}

// B fragment 32x16 bf16 from B^T stored row-major [n][k] (k contiguous), pre-offset.
__device__ __forceinline__ v16bf frag_b(const bf16_t* srcT, int ld, int half, int l16) {
  const bf16_t* p = srcT + (size_t)l16 * ld + half * 8;
  return mk16(p, p + 16);
}

__device__ __forceinline__ v8f wmma_bf16(v16bf a, v16bf b, v8f c) {
  return __builtin_amdgcn_wmma_f32_16x16x32_bf16(false, a, false, b, (short)0, c,
                                                 false, false);
}

// ---------------------------------------------------------------------------
// Kernel 1: 1x1 conv projection (GEMM W[128x128] @ X[128x64] per tile) + epilogue
// mode 0: per-pixel mean-subtract + L2 normalize, store [p][c] bf16
// mode 1: plain, store [c][p] bf16
// ---------------------------------------------------------------------------
__global__ __launch_bounds__(256) void proj_kernel(
    const float* __restrict__ img,   // [B][C][HW]
    const float* __restrict__ Wm,    // [C][C] row-major (o, c)
    const float* __restrict__ bias,  // [C]
    bf16_t* __restrict__ outp, int mode)
{
  const int b = blockIdx.y;
  const int p0 = blockIdx.x * 64;
  const int tid = threadIdx.x;
  const int wave = tid >> 5, lane = tid & 31;
  const int half = lane >> 4, l16 = lane & 15;

  __shared__ alignas(16) bf16_t Xs[64][136];   // [p][c] bf16 (padded)
  __shared__ alignas(16) float  Ys[64][132];   // [p][c] f32  (padded)

  const float* imgb = img + (size_t)b * Cn * HWn;

  // Load X tile [128 c][64 p] -> Xs[p][c] (bf16, transposed)
  for (int i = tid; i < 128 * 16; i += 256) {
    int c = i >> 4, q = i & 15;
    float4 v = *(const float4*)(imgb + (size_t)c * HWn + p0 + q * 4);
    Xs[q * 4 + 0][c] = f2bf(v.x);
    Xs[q * 4 + 1][c] = f2bf(v.y);
    Xs[q * 4 + 2][c] = f2bf(v.z);
    Xs[q * 4 + 3][c] = f2bf(v.w);
  }
  __syncthreads();

  // W fragments for this wave's 16 output channels, 4 K-chunks of 32
  const int c0 = wave * 16;
  v16bf afrag[4];
#pragma unroll
  for (int kc = 0; kc < 4; ++kc) {
    const float* wp = Wm + (size_t)(c0 + l16) * Cn + kc * 32;
    v16bf r;
#pragma unroll
    for (int e = 0; e < 8; ++e) {
      r[e]     = f2bf(wp[half * 8 + e]);
      r[e + 8] = f2bf(wp[16 + half * 8 + e]);
    }
    afrag[kc] = r;
  }

  v8f acc[4];
#pragma unroll
  for (int s = 0; s < 4; ++s) acc[s] = (v8f)0.f;

#pragma unroll
  for (int kc = 0; kc < 4; ++kc) {
#pragma unroll
    for (int s = 0; s < 4; ++s) {
      v16bf bf = frag_b(&Xs[s * 16][kc * 32], 136, half, l16);
      acc[s] = wmma_bf16(afrag[kc], bf, acc[s]);
    }
  }

  // bias (row M = i + 8*half) and stash to Ys[p][c]
  float bl[8];
#pragma unroll
  for (int i = 0; i < 8; ++i) bl[i] = bias[c0 + 8 * half + i];
#pragma unroll
  for (int s = 0; s < 4; ++s)
#pragma unroll
    for (int i = 0; i < 8; ++i)
      Ys[s * 16 + l16][c0 + 8 * half + i] = acc[s][i] + bl[i];
  __syncthreads();

  if (mode == 0) {
    if (tid < 64) {
      int p = tid;
      float sum = 0.f;
#pragma unroll 8
      for (int c = 0; c < Cn; ++c) sum += Ys[p][c];
      float mean = sum * (1.0f / Cn);
      float ss = 0.f;
#pragma unroll 8
      for (int c = 0; c < Cn; ++c) { float d = Ys[p][c] - mean; ss += d * d; }
      float inv = rsqrtf(ss);
      bf16_t* op = outp + ((size_t)b * HWn + p0 + p) * Cn;
#pragma unroll 8
      for (int c = 0; c < Cn; ++c) op[c] = f2bf((Ys[p][c] - mean) * inv);
    }
  } else {
    for (int i = tid; i < 128 * 64; i += 256) {
      int c = i >> 6, p = i & 63;
      outp[(size_t)b * Cn * HWn + (size_t)c * HWn + p0 + p] = f2bf(Ys[p][c]);
    }
  }
}

// ---------------------------------------------------------------------------
// Kernel 2: corr[b][n][m] = pq[b][n][:] . pkT[b][m][:]   (128x128 tiles, K=128)
// ---------------------------------------------------------------------------
__global__ __launch_bounds__(256) void corr_kernel(
    const bf16_t* __restrict__ pq,   // [B][HW][C]
    const bf16_t* __restrict__ pkT,  // [B][HW][C]
    float* __restrict__ corr)        // [B][HW][HW]
{
  const int b = blockIdx.z;
  const int n0 = blockIdx.y * 128, m0 = blockIdx.x * 128;
  const int tid = threadIdx.x;
  const int wave = tid >> 5, lane = tid & 31;
  const int half = lane >> 4, l16 = lane & 15;

  __shared__ alignas(16) bf16_t Bs[128][136];

  // Cooperative stage of pkT tile [m0..+128][0..128] into LDS
  const bf16_t* kb = pkT + ((size_t)b * HWn + m0) * Cn;
  for (int i = tid; i < 128 * 16; i += 256) {
    int m = i >> 4, q = i & 15;
    *(v8bf*)(&Bs[m][q * 8]) = *(const v8bf*)(kb + (size_t)m * Cn + q * 8);
  }
  __syncthreads();

  const bf16_t* qb = pq + ((size_t)b * HWn + n0 + wave * 16) * Cn;
  v16bf a[4];
#pragma unroll
  for (int kc = 0; kc < 4; ++kc) a[kc] = frag_a(qb + kc * 32, Cn, half, l16);

  v8f acc[8];
#pragma unroll
  for (int s = 0; s < 8; ++s) acc[s] = (v8f)0.f;

#pragma unroll
  for (int kc = 0; kc < 4; ++kc) {
#pragma unroll
    for (int s = 0; s < 8; ++s) {
      v16bf bf = frag_b(&Bs[s * 16][kc * 32], 136, half, l16);
      acc[s] = wmma_bf16(a[kc], bf, acc[s]);
    }
  }

  float* cb = corr + ((size_t)b * HWn + n0 + wave * 16) * HWn + m0;
#pragma unroll
  for (int s = 0; s < 8; ++s)
#pragma unroll
    for (int i = 0; i < 8; ++i)
      cb[(size_t)(i + 8 * half) * HWn + s * 16 + l16] = acc[s][i];
}

// ---------------------------------------------------------------------------
// Kernel 3: per-row online softmax stats; conf = rowmax; rowsum = sum exp((x-mx)/tau)
// ---------------------------------------------------------------------------
__global__ __launch_bounds__(256) void stats_kernel(
    const float* __restrict__ corr, const float* __restrict__ tau,
    float* __restrict__ conf, float* __restrict__ rowmax, float* __restrict__ rowsum)
{
  const int r = blockIdx.x;
  const int tid = threadIdx.x;
  const float* row = corr + (size_t)r * HWn;
  const float inv_tau = 1.0f / tau[0];

  float mx = -INFINITY, sm = 0.f;
  for (int j = tid; j < HWn / 4; j += 256) {
    float4 v = *(const float4*)(row + j * 4);
    float vv[4] = {v.x, v.y, v.z, v.w};
#pragma unroll
    for (int e = 0; e < 4; ++e) {
      float f = vv[e];
      float m2 = fmaxf(mx, f);
      sm = sm * __expf((mx - m2) * inv_tau) + __expf((f - m2) * inv_tau);
      mx = m2;
    }
  }

  __shared__ float smx[256], ssm[256];
  smx[tid] = mx; ssm[tid] = sm;
  __syncthreads();
  for (int off = 128; off > 0; off >>= 1) {
    if (tid < off) {
      float m1 = smx[tid], s1 = ssm[tid];
      float m2 = smx[tid + off], s2 = ssm[tid + off];
      float m = fmaxf(m1, m2);
      smx[tid] = m;
      ssm[tid] = s1 * __expf((m1 - m) * inv_tau) + s2 * __expf((m2 - m) * inv_tau);
    }
    __syncthreads();
  }
  if (tid == 0) { conf[r] = smx[0]; rowmax[r] = smx[0]; rowsum[r] = ssm[0]; }
}

// ---------------------------------------------------------------------------
// Kernel 4: out[b][c][n] = gamma * (sum_m pv[b][c][m] * e[n][m]) / rowsum[n] + value
// flash-style: stream corr tiles, exponentiate to bf16 in LDS (double-buffered,
// one barrier per k-step; staging of tile k+1 overlaps WMMAs on tile k).
// n-tile = 64 -> 256 workgroups for full memory-system occupancy.
// ---------------------------------------------------------------------------
__global__ __launch_bounds__(256) void out_kernel(
    const bf16_t* __restrict__ pv,   // [B][C][HW]
    const float* __restrict__ corr,
    const float* __restrict__ rowmax, const float* __restrict__ rowsum,
    const float* __restrict__ value_img, const float* __restrict__ gamma,
    const float* __restrict__ tau,
    float* __restrict__ outp)        // [B][C][HW]
{
  const int b = blockIdx.y;
  const int n0 = blockIdx.x * 64;
  const int tid = threadIdx.x;
  const int wave = tid >> 5, lane = tid & 31;
  const int half = lane >> 4, l16 = lane & 15;

  __shared__ alignas(16) bf16_t Es[2][64][40];   // ping-pong exp tiles [n][m32]
  __shared__ float rmx[64];

  const float inv_tau = 1.0f / tau[0];
  if (tid < 64) rmx[tid] = rowmax[b * HWn + n0 + tid];

  const bf16_t* ab = pv + ((size_t)b * Cn + wave * 16) * HWn;
  const float* cb = corr + ((size_t)b * HWn + n0) * HWn;

  // each thread stages 2 of the 512 (row, quad) chunks of a 64x32 tile
  auto stage = [&](int buf, int m) {
#pragma unroll
    for (int j = 0; j < 2; ++j) {
      int i = tid + 256 * j;
      int n = i >> 3, q = i & 7;
      float4 v = *(const float4*)(cb + (size_t)n * HWn + m + q * 4);
      float rm = rmx[n];
      Es[buf][n][q * 4 + 0] = f2bf(__expf((v.x - rm) * inv_tau));
      Es[buf][n][q * 4 + 1] = f2bf(__expf((v.y - rm) * inv_tau));
      Es[buf][n][q * 4 + 2] = f2bf(__expf((v.z - rm) * inv_tau));
      Es[buf][n][q * 4 + 3] = f2bf(__expf((v.w - rm) * inv_tau));
    }
  };

  v8f acc[4];
#pragma unroll
  for (int s = 0; s < 4; ++s) acc[s] = (v8f)0.f;

  __syncthreads();          // rmx visible
  stage(0, 0);              // prologue
  __syncthreads();

  const int NIT = HWn / 32;
  for (int it = 0; it < NIT; ++it) {
    const int cur = it & 1;
    if (it + 1 < NIT) stage(cur ^ 1, (it + 1) * 32);   // overlap with WMMAs below

    v16bf a = frag_a(ab + it * 32, HWn, half, l16);
#pragma unroll
    for (int s = 0; s < 4; ++s) {
      v16bf bf = frag_b(&Es[cur][s * 16][0], 40, half, l16);
      acc[s] = wmma_bf16(a, bf, acc[s]);
    }
    __syncthreads();        // readers of buf `cur` done; writers of buf `cur^1` done
  }

  const float g = gamma[0];
#pragma unroll
  for (int s = 0; s < 4; ++s) {
    int n = n0 + s * 16 + l16;
    float inv_rs = g / rowsum[b * HWn + n];
#pragma unroll
    for (int i = 0; i < 8; ++i) {
      int c = wave * 16 + i + 8 * half;
      size_t idx = ((size_t)b * Cn + c) * HWn + n;
      outp[idx] = acc[s][i] * inv_rs + value_img[idx];
    }
  }
}

// ---------------------------------------------------------------------------
extern "C" void kernel_launch(void* const* d_in, const int* in_sizes, int n_in,
                              void* d_out, int out_size, void* d_ws, size_t ws_size,
                              hipStream_t stream) {
  (void)in_sizes; (void)n_in; (void)out_size; (void)ws_size;
  const float* key_img   = (const float*)d_in[0];
  const float* query_img = (const float*)d_in[1];
  const float* value_img = (const float*)d_in[2];
  const float* Wq = (const float*)d_in[3];
  const float* bq = (const float*)d_in[4];
  const float* Wk = (const float*)d_in[5];
  const float* bk = (const float*)d_in[6];
  const float* Wv = (const float*)d_in[7];
  const float* bv = (const float*)d_in[8];
  const float* gamma = (const float*)d_in[9];
  const float* tau   = (const float*)d_in[10];

  float* out  = (float*)d_out;
  float* corr = out;                                        // [B][HW][HW]
  float* conf = out + (size_t)Bn * HWn * HWn;               // [B][HW]
  float* outt = conf + (size_t)Bn * HWn;                    // [B][C][HW]

  bf16_t* pq  = (bf16_t*)d_ws;                              // [B][HW][C]
  bf16_t* pkT = pq  + (size_t)Bn * HWn * Cn;                // [B][HW][C]
  bf16_t* pv  = pkT + (size_t)Bn * HWn * Cn;                // [B][C][HW]
  float* rowmax = (float*)(pv + (size_t)Bn * HWn * Cn);     // [B*HW]
  float* rowsum = rowmax + (size_t)Bn * HWn;                // [B*HW]

  dim3 blk(256);
  proj_kernel<<<dim3(HWn / 64, Bn), blk, 0, stream>>>(query_img, Wq, bq, pq, 0);
  proj_kernel<<<dim3(HWn / 64, Bn), blk, 0, stream>>>(key_img,   Wk, bk, pkT, 0);
  proj_kernel<<<dim3(HWn / 64, Bn), blk, 0, stream>>>(value_img, Wv, bv, pv, 1);
  corr_kernel<<<dim3(HWn / 128, HWn / 128, Bn), blk, 0, stream>>>(pq, pkT, corr);
  stats_kernel<<<dim3(Bn * HWn), blk, 0, stream>>>(corr, tau, conf, rowmax, rowsum);
  out_kernel<<<dim3(HWn / 64, Bn), blk, 0, stream>>>(pv, corr, rowmax, rowsum,
                                                     value_img, gamma, tau, outt);
}
